// GraphConstructor_13134009991523
// MI455X (gfx1250) — matrix operand order repaired and checked by reading, not compiled, not run on hardware
//
#include <hip/hip_runtime.h>
#include <hip/hip_bf16.h>

#define N_NODES   2000
#define NFEAT     128
#define LOOKBACK  128
#define TIMEF     36
#define FC1       256
#define EMB_COLS  (N_NODES * NFEAT)   // 256000
#define DELTA_MIN 0.05f

typedef __attribute__((ext_vector_type(16))) __bf16 bf16x16;
typedef __attribute__((ext_vector_type(8)))  float  floatx8;

static __device__ __forceinline__ __bf16 f2bf(float f) { return (__bf16)f; }

static __device__ __forceinline__ floatx8 wmma_bf16(bf16x16 a, bf16x16 b, floatx8 c) {
  // D = A(16x32 bf16) * B(32x16 bf16) + C(16x16 f32)
  return __builtin_amdgcn_wmma_f32_16x16x32_bf16(false, a, false, b, (short)0, c,
                                                 false, false);
}

// Fragment index maps (ISA 7.12.2, wave32):
//  A 16x32 bf16 : lane m = lane&15, element e -> k = (e/8)*16 + (lane/16)*8 + e%8
//  B 32x16 bf16 : lane n = lane&15, element e -> k = (e/8)*16 + (lane/16)*8 + e%8
//  C 16x16 f32  : lane n = lane&15, vgpr v    -> m = v + 8*(lane/16)
#define AKIDX(e, kg) ((((e) >> 3) << 4) + ((kg) << 3) + ((e) & 7))
#define BKIDX(e, kg) ((((e) >> 3) << 4) + ((kg) << 3) + ((e) & 7))
#define ZERO8 {0.f,0.f,0.f,0.f,0.f,0.f,0.f,0.f}

// Load 8 contiguous floats -> fragment elements [base, base+8) (two b128 loads)
static __device__ __forceinline__ void loadf8(bf16x16& f, int base, const float* p) {
  float4 u = *(const float4*)(p);
  float4 v = *(const float4*)(p + 4);
  f[base + 0] = f2bf(u.x); f[base + 1] = f2bf(u.y);
  f[base + 2] = f2bf(u.z); f[base + 3] = f2bf(u.w);
  f[base + 4] = f2bf(v.x); f[base + 5] = f2bf(v.y);
  f[base + 6] = f2bf(v.z); f[base + 7] = f2bf(v.w);
}

// ---------------------------------------------------------------------------
// Kernel 1: H_s = relu(layer_initial @ Ws1 + bs1)  (128x256, K=2000)
//           H_t = relu(time_feats   @ Wt1 + bt1)  (128x256, K=36)
// One wave per 16x16 tile; 256 waves. Branch-free clamped tail.
// ---------------------------------------------------------------------------
__global__ __launch_bounds__(128) void k_hidden(
    const float* __restrict__ Lmat, const float* __restrict__ Tmat,
    const float* __restrict__ Ws1,  const float* __restrict__ bs1,
    const float* __restrict__ Wt1,  const float* __restrict__ bt1,
    float* __restrict__ Hs, float* __restrict__ Ht)
{
  const int wave = blockIdx.x * 4 + (threadIdx.x >> 5);
  const int lane = threadIdx.x & 31;
  const bool spatial = wave < 128;
  const int tile = spatial ? wave : wave - 128;
  const int tm = tile >> 4;      // row-block of T
  const int tn = tile & 15;      // col-block of FC1
  const float* A  = spatial ? Lmat : Tmat;
  const float* Wp = spatial ? Ws1  : Wt1;
  const float* bp = spatial ? bs1  : bt1;
  float*       Hp = spatial ? Hs   : Ht;
  const int K     = spatial ? N_NODES : TIMEF;
  const int kmain = K & ~31;     // 1984 / 32

  const int mA = lane & 15;
  const int kg = (lane >> 4) & 1;
  const int nB = lane & 15;
  const int col = tn * 16 + nB;
  const float* arow = A + (tm * 16 + mA) * K;

  floatx8 acc = ZERO8;
  for (int k0 = 0; k0 < kmain; k0 += 32) {
    bf16x16 a, b;
    loadf8(a, 0, arow + k0 + (kg << 3));
    loadf8(a, 8, arow + k0 + 16 + (kg << 3));
#pragma unroll
    for (int e = 0; e < 16; ++e)
      b[e] = f2bf(Wp[(k0 + BKIDX(e, kg)) * FC1 + col]);
    acc = wmma_bf16(a, b, acc);
  }
  { // tail block (K % 32 != 0 for both paths): clamp+select, no branches
    bf16x16 a, b;
#pragma unroll
    for (int e = 0; e < 16; ++e) {
      const int ka = kmain + AKIDX(e, kg);
      const float va = arow[ka < K ? ka : 0];
      a[e] = f2bf(ka < K ? va : 0.f);
      const int kb = kmain + BKIDX(e, kg);
      const float vb = Wp[(kb < K ? kb : 0) * FC1 + col];
      b[e] = f2bf(kb < K ? vb : 0.f);
    }
    acc = wmma_bf16(a, b, acc);
  }

  const float bia = bp[col];
#pragma unroll
  for (int v = 0; v < 8; ++v) {
    const int row = tm * 16 + v + (kg << 3);
    float val = acc[v] + bia;
    Hp[row * FC1 + col] = val > 0.f ? val : 0.f;
  }
}

// ---------------------------------------------------------------------------
// Kernel 2 (dominant): gathered 2nd layer, U1 and U2 fused in one wave so the
// i1/i2 column gathers share cachelines (|i1-i2|*4 bytes apart).
//   U1[node][t] = relu(Hs[t,:]·Ws2[:,node*128+i1]+bs2) + relu(Ht·Wt2+bt2)
// Output node-major (N_NODES x LOOKBACK). 8*125 = 1000 waves.
// ---------------------------------------------------------------------------
__global__ __launch_bounds__(128) void k_embed(
    const float* __restrict__ Hs,  const float* __restrict__ Ht,
    const float* __restrict__ Ws2, const float* __restrict__ bs2,
    const float* __restrict__ Wt2, const float* __restrict__ bt2,
    const int* __restrict__ idxp,  const int* __restrict__ tdp,
    float* __restrict__ U1, float* __restrict__ U2)
{
  const int idx = idxp[0];
  const int td  = tdp[0];
  const int i1  = (td >= 0) ? (idx - td) : idx;
  const int i2  = (td >= 0) ? idx        : (idx + td);

  const int wave = blockIdx.x * 4 + (threadIdx.x >> 5);
  const int lane = threadIdx.x & 31;
  const int tm = wave / 125;   // t block    (0..7)
  const int tn = wave % 125;   // node block (0..124)
  const int mA = lane & 15;
  const int kg = (lane >> 4) & 1;
  const int nB = lane & 15;

  const float* hsrow = Hs + (tm * 16 + mA) * FC1;
  const float* htrow = Ht + (tm * 16 + mA) * FC1;
  const int    node  = tn * 16 + nB;
  const size_t col1  = (size_t)node * NFEAT + i1;
  const size_t col2  = (size_t)node * NFEAT + i2;

  floatx8 aS1 = ZERO8, aS2 = ZERO8, aT1 = ZERO8, aT2 = ZERO8;
#pragma unroll 1
  for (int k0 = 0; k0 < FC1; k0 += 32) {
    bf16x16 fS, fT, bS1, bS2, bT1, bT2;
    loadf8(fS, 0, hsrow + k0 + (kg << 3));
    loadf8(fS, 8, hsrow + k0 + 16 + (kg << 3));
    loadf8(fT, 0, htrow + k0 + (kg << 3));
    loadf8(fT, 8, htrow + k0 + 16 + (kg << 3));
#pragma unroll
    for (int e = 0; e < 16; ++e) {
      const size_t rowoff = (size_t)(k0 + BKIDX(e, kg)) * EMB_COLS;
      bS1[e] = f2bf(Ws2[rowoff + col1]);   // same cacheline as col2 (usually)
      bS2[e] = f2bf(Ws2[rowoff + col2]);
      bT1[e] = f2bf(Wt2[rowoff + col1]);
      bT2[e] = f2bf(Wt2[rowoff + col2]);
    }
    aS1 = wmma_bf16(fS, bS1, aS1);
    aS2 = wmma_bf16(fS, bS2, aS2);
    aT1 = wmma_bf16(fT, bT1, aT1);
    aT2 = wmma_bf16(fT, bT2, aT2);
  }

  const float bS1v = bs2[col1], bS2v = bs2[col2];
  const float bT1v = bt2[col1], bT2v = bt2[col2];
  const int tbase = tm * 16 + (kg << 3);   // 8 contiguous t per lane
  float4 lo1, hi1, lo2, hi2;
#pragma unroll
  for (int v = 0; v < 8; ++v) {
    float s1 = aS1[v] + bS1v; s1 = s1 > 0.f ? s1 : 0.f;
    float q1 = aT1[v] + bT1v; q1 = q1 > 0.f ? q1 : 0.f;
    float s2 = aS2[v] + bS2v; s2 = s2 > 0.f ? s2 : 0.f;
    float q2 = aT2[v] + bT2v; q2 = q2 > 0.f ? q2 : 0.f;
    const float o1 = s1 + q1, o2 = s2 + q2;
    if (v == 0) { lo1.x = o1; lo2.x = o2; }
    if (v == 1) { lo1.y = o1; lo2.y = o2; }
    if (v == 2) { lo1.z = o1; lo2.z = o2; }
    if (v == 3) { lo1.w = o1; lo2.w = o2; }
    if (v == 4) { hi1.x = o1; hi2.x = o2; }
    if (v == 5) { hi1.y = o1; hi2.y = o2; }
    if (v == 6) { hi1.z = o1; hi2.z = o2; }
    if (v == 7) { hi1.w = o1; hi2.w = o2; }
  }
  float* u1p = U1 + (size_t)node * LOOKBACK + tbase;
  float* u2p = U2 + (size_t)node * LOOKBACK + tbase;
  *(float4*)(u1p)     = lo1;
  *(float4*)(u1p + 4) = hi1;
  *(float4*)(u2p)     = lo2;
  *(float4*)(u2p + 4) = hi2;
}

// ---------------------------------------------------------------------------
// Kernel 3: W = U1 @ B   (2000x128 @ 128x128), U1 node-major. 1000 waves.
// ---------------------------------------------------------------------------
__global__ __launch_bounds__(128) void k_u1b(
    const float* __restrict__ U1, const float* __restrict__ Bm,
    float* __restrict__ W)
{
  const int wave = blockIdx.x * 4 + (threadIdx.x >> 5);
  const int lane = threadIdx.x & 31;
  const int tm = wave >> 3;    // node block (0..124)
  const int tj = wave & 7;     // feature block (0..7)
  const int mA = lane & 15;
  const int kg = (lane >> 4) & 1;
  const int nB = lane & 15;

  const float* u1row = U1 + (size_t)(tm * 16 + mA) * LOOKBACK;  // A[m=node][k=t]
  floatx8 acc = ZERO8;
#pragma unroll
  for (int k0 = 0; k0 < LOOKBACK; k0 += 32) {
    bf16x16 a, b;
    loadf8(a, 0, u1row + k0 + (kg << 3));
    loadf8(a, 8, u1row + k0 + 16 + (kg << 3));
#pragma unroll
    for (int e = 0; e < 16; ++e)
      b[e] = f2bf(Bm[(k0 + BKIDX(e, kg)) * NFEAT + tj * 16 + nB]);
    acc = wmma_bf16(a, b, acc);
  }
#pragma unroll
  for (int v = 0; v < 8; ++v) {
    const int node = tm * 16 + v + (kg << 3);
    W[node * NFEAT + tj * 16 + nB] = acc[v];
  }
}

// ---------------------------------------------------------------------------
// Kernel 4: x = W @ U2^T, fused threshold. U2 node-major so B-fragment loads
// are per-lane contiguous b128. 125*125 = 15625 waves.
// ---------------------------------------------------------------------------
__global__ __launch_bounds__(128) void k_scores(
    const float* __restrict__ W, const float* __restrict__ U2,
    float* __restrict__ X)
{
  const int wave = blockIdx.x * 4 + (threadIdx.x >> 5);
  if (wave >= 125 * 125) return;
  const int lane = threadIdx.x & 31;
  const int ti = wave / 125;   // row-node block
  const int tj = wave % 125;   // col-node block
  const int mA = lane & 15;
  const int kg = (lane >> 4) & 1;
  const int nB = lane & 15;

  const float* wrow  = W  + (size_t)(ti * 16 + mA) * NFEAT;   // A[m][k=j]
  const float* u2row = U2 + (size_t)(tj * 16 + nB) * NFEAT;   // B[k=j][n] = U2[n][j]
  floatx8 acc = ZERO8;
#pragma unroll
  for (int k0 = 0; k0 < NFEAT; k0 += 32) {
    bf16x16 a, b;
    loadf8(a, 0, wrow + k0 + (kg << 3));
    loadf8(a, 8, wrow + k0 + 16 + (kg << 3));
    loadf8(b, 0, u2row + k0 + (kg << 3));       // two runs of 8 floats, same k as A
    loadf8(b, 8, u2row + k0 + 16 + (kg << 3));
    acc = wmma_bf16(a, b, acc);
  }
#pragma unroll
  for (int v = 0; v < 8; ++v) {
    const int row = ti * 16 + v + (kg << 3);
    const int col = tj * 16 + nB;
    const float x = acc[v];
    X[(size_t)row * N_NODES + col] = (x >= DELTA_MIN) ? x : 0.f;
  }
}

// ---------------------------------------------------------------------------
// Kernel 5: in-place row softmax over 2000 columns. One block per row.
// ---------------------------------------------------------------------------
__global__ __launch_bounds__(256) void k_softmax(float* __restrict__ X)
{
  __shared__ float red[256];
  const int tid = threadIdx.x;
  float* rp = X + (size_t)blockIdx.x * N_NODES;

  float mx = -3.402823466e38f;
  for (int c = tid; c < N_NODES; c += 256) mx = fmaxf(mx, rp[c]);
  red[tid] = mx; __syncthreads();
  for (int s = 128; s > 0; s >>= 1) {
    if (tid < s) red[tid] = fmaxf(red[tid], red[tid + s]);
    __syncthreads();
  }
  const float M = red[0];
  __syncthreads();

  float sum = 0.f;
  for (int c = tid; c < N_NODES; c += 256) sum += __expf(rp[c] - M);
  red[tid] = sum; __syncthreads();
  for (int s = 128; s > 0; s >>= 1) {
    if (tid < s) red[tid] += red[tid + s];
    __syncthreads();
  }
  const float inv = 1.f / red[0];
  __syncthreads();

  for (int c = tid; c < N_NODES; c += 256) rp[c] = __expf(rp[c] - M) * inv;
}

// ---------------------------------------------------------------------------
extern "C" void kernel_launch(void* const* d_in, const int* in_sizes, int n_in,
                              void* d_out, int out_size, void* d_ws, size_t ws_size,
                              hipStream_t stream) {
  const float* time_feats    = (const float*)d_in[0];
  const float* layer_initial = (const float*)d_in[1];
  const float* Ws1 = (const float*)d_in[2];
  const float* bs1 = (const float*)d_in[3];
  const float* Ws2 = (const float*)d_in[4];
  const float* bs2 = (const float*)d_in[5];
  const float* Wt1 = (const float*)d_in[6];
  const float* bt1 = (const float*)d_in[7];
  const float* Wt2 = (const float*)d_in[8];
  const float* bt2 = (const float*)d_in[9];
  const float* Bm  = (const float*)d_in[10];
  const int*   idx = (const int*)d_in[11];
  const int*   td  = (const int*)d_in[12];
  float* out = (float*)d_out;

  // Workspace (floats): Hs 32768 | Ht 32768 | U1 256000 | U2 256000 | W 256000
  float* ws   = (float*)d_ws;
  float* Hs   = ws;
  float* Ht   = Hs + LOOKBACK * FC1;
  float* U1   = Ht + LOOKBACK * FC1;
  float* U2   = U1 + N_NODES * LOOKBACK;
  float* Wbuf = U2 + N_NODES * LOOKBACK;

  k_hidden <<<64,  128, 0, stream>>>(layer_initial, time_feats, Ws1, bs1, Wt1, bt1, Hs, Ht);
  k_embed  <<<250, 128, 0, stream>>>(Hs, Ht, Ws2, bs2, Wt2, bt2, idx, td, U1, U2);
  k_u1b    <<<250, 128, 0, stream>>>(U1, Bm, Wbuf);
  k_scores <<<(125 * 125 + 3) / 4, 128, 0, stream>>>(Wbuf, U2, out);
  k_softmax<<<N_NODES, 256, 0, stream>>>(out);
}
